// MLAAttention_49959059587361
// MI455X (gfx1250) — matrix operand order repaired and checked by reading, not compiled, not run on hardware
//
#include <hip/hip_runtime.h>

// MLA attention for MI455X (gfx1250), wave32, WMMA bf16 16x16x32.
// Pass 0: f32 -> bf16 conversion of ckv / w_up / rope_k (one-time, off hot path).
// Pass 1: kv = ckv @ w_up^T on WMMA; outputs k_nope [T,H*128] bf16 row-major and
//         V transposed [B,H,128,S] bf16 (keys contiguous for the p@v B-matrix).
// Pass 2: causal flash attention, all contractions on WMMA, online softmax via
//         wave32 shuffles, P relayout through LDS as bf16, V tiles double-
//         buffered through LDS with global_load_async_to_lds_b128 (ASYNCcnt).

typedef __attribute__((ext_vector_type(16))) __bf16 v16bf;
typedef __attribute__((ext_vector_type(8)))  float  v8f;

#define SEQ     1024
#define BATCH   4
#define HEADS   16
#define NOPE    128
#define ROPE    64
#define VDIM    128
#define RANK    512
#define T_TOK   (BATCH*SEQ)
#define SCALE   0.07216878364870323f
#define LOG2E   1.44269504088896f
#define SM_SCALE (SCALE * LOG2E)      // fold softmax into exp2 domain
#define NEG_BIG (-3.0e38f)

union BFrag {
  v16bf v;
  unsigned u32[8];
  uint4 q4[2];
};

static __device__ __forceinline__ unsigned short bf_bits(float x) {
  union { __bf16 b; unsigned short s; } o; o.b = (__bf16)x; return o.s;
}
static __device__ __forceinline__ unsigned pack2bf(float lo, float hi) {
  union { __bf16 b[2]; unsigned u; } o;
  o.b[0] = (__bf16)lo; o.b[1] = (__bf16)hi;
  return o.u;
}
// 16-bit A-matrix (16xK) per-lane K offset (contiguous order):
// VGPR v -> 2v, plus 16 for lanes 16-31.
static __device__ __forceinline__ int a_koff(int v, int half) {
  return 2*v + half*16;
}
// Async copy 16B global -> LDS, per-lane addresses, tracked by ASYNCcnt.
static __device__ __forceinline__ void async_g2l_b128(unsigned lds_addr,
                                                      const void* gptr) {
  const unsigned long long ga = (unsigned long long)(size_t)gptr;
  asm volatile("global_load_async_to_lds_b128 %0, %1, off"
               :: "v"(lds_addr), "v"(ga) : "memory");
}

// ---------------------------------------------------------------------------
// Pass 0: bulk f32 -> bf16 (4 elements/thread, hardware cvt).
// ---------------------------------------------------------------------------
__global__ __launch_bounds__(256)
void cvt_bf16(const float* __restrict__ src, unsigned short* __restrict__ dst, int n4)
{
  const int i = blockIdx.x * 256 + threadIdx.x;
  if (i < n4) {
    const float4 f = ((const float4*)src)[i];
    union { __bf16 b[4]; uint2 u; } o;
    o.b[0] = (__bf16)f.x; o.b[1] = (__bf16)f.y;
    o.b[2] = (__bf16)f.z; o.b[3] = (__bf16)f.w;
    ((uint2*)dst)[i] = o.u;
  }
}

// ---------------------------------------------------------------------------
// Pass 1: up-projection GEMM. M=T=4096, N=4096, K=512, bf16 in / f32 acc.
// Block = 128 threads (4 waves); wave w -> rows [by*64+16w,+16), cols [bx*64,+64).
// ---------------------------------------------------------------------------
__global__ __launch_bounds__(128)
void mla_upproj(const unsigned short* __restrict__ ckv_bf,
                const unsigned short* __restrict__ wup_bf,
                unsigned short* __restrict__ k_ws,
                unsigned short* __restrict__ vT_ws)
{
  const int lane = threadIdx.x & 31;
  const int wave = threadIdx.x >> 5;
  const int half = lane >> 4;
  const int l16  = lane & 15;
  const int rowbase = blockIdx.y * 64 + wave * 16;
  const int nbase   = blockIdx.x * 64;

  const v8f vz = {0.f,0.f,0.f,0.f,0.f,0.f,0.f,0.f};
  v8f acc[4];
#pragma unroll
  for (int i = 0; i < 4; ++i) acc[i] = vz;

  const unsigned short* arow = ckv_bf + (size_t)(rowbase + l16) * RANK;

#pragma unroll
  for (int kc = 0; kc < RANK; kc += 32) {
    BFrag a;                 // per-lane data is 2x16 contiguous bf16
    a.q4[0] = *(const uint4*)(arow + kc + half*16);
    a.q4[1] = *(const uint4*)(arow + kc + half*16 + 8);
#pragma unroll
    for (int nt = 0; nt < 4; ++nt) {
      // B[k][n] = w_up[nbase+nt*16+n][kc+k]; lane col=l16, K rows half*16+0..15
      const unsigned short* bcol =
          wup_bf + (size_t)(nbase + nt*16 + l16) * RANK + kc + half*16;
      BFrag bm;
      bm.q4[0] = *(const uint4*)(bcol);
      bm.q4[1] = *(const uint4*)(bcol + 8);
      acc[nt] = __builtin_amdgcn_wmma_f32_16x16x32_bf16(
          false, a.v, false, bm.v, (short)0, acc[nt], false, false);
    }
  }

  // C layout: lane -> (row = v + 8*half, col = l16). Split k_nope / V^T stores.
#pragma unroll
  for (int nt = 0; nt < 4; ++nt) {
#pragma unroll
    for (int v = 0; v < 8; ++v) {
      const int t = rowbase + v + 8*half;
      const int j = nbase + nt*16 + l16;
      const int h = j >> 8;
      const int d = j & 255;
      const unsigned short bits = bf_bits(acc[nt][v]);
      if (d < NOPE) {
        k_ws[(size_t)t * (HEADS*NOPE) + h*NOPE + d] = bits;
      } else {
        const int bb = t >> 10, ss = t & (SEQ-1);
        vT_ws[(((size_t)(bb*HEADS + h) * VDIM) + (d - NOPE)) * SEQ + ss] = bits;
      }
    }
  }
}

// ---------------------------------------------------------------------------
// Pass 2: causal flash attention. Block = 4 waves; wave owns 16 q-rows.
// Grid: (SEQ/64, HEADS, BATCH).
// ---------------------------------------------------------------------------
__global__ __launch_bounds__(128)
void mla_attn(const float* __restrict__ q,
              const unsigned short* __restrict__ rope_bf,
              const unsigned short* __restrict__ k_ws,
              const unsigned short* __restrict__ vT_ws,
              float* __restrict__ out)
{
  __shared__ unsigned short pbuf[4][16*32];        // P tile as bf16, per wave
  __shared__ unsigned short vstage[4][2][8*32*16]; // V tile dbl-buf, per wave
  const int lane = threadIdx.x & 31;
  const int wave = threadIdx.x >> 5;
  const int half = lane >> 4;
  const int l16  = lane & 15;
  const int h  = blockIdx.y;
  const int b  = blockIdx.z;
  const int qb = blockIdx.x * 64 + wave * 16;
  unsigned short* pw = pbuf[wave];

  const unsigned short* vbase =
      vT_ws + ((size_t)(b*HEADS + h) * VDIM) * SEQ;

  // Issue the async V-tile copy for key tile jj into LDS buffer `buf`.
  // Each lane copies the exact 32B it will consume per nt (no cross-lane).
  auto v_prefetch = [&](int jj, int buf) {
#pragma unroll
    for (int nt = 0; nt < 8; ++nt) {
      const unsigned short* g = vbase + (size_t)(nt*16 + l16) * SEQ + jj + half*16;
      unsigned short* slot = &vstage[wave][buf][(nt*32 + lane) * 16];
      const unsigned lds0 = (unsigned)(size_t)slot;
      async_g2l_b128(lds0, g);            // bytes 0..15  (LDS and MEM share offset)
      asm volatile("global_load_async_to_lds_b128 %0, %1, off offset:16"
                   :: "v"(lds0), "v"((unsigned long long)(size_t)g) : "memory");
    }
  };

  // Q fragments (A layout), f32 -> bf16 once per wave
  BFrag qn[4], qr[2];
  {
    const int tq = b * SEQ + qb + l16;
    const float* qrow = q + ((size_t)tq * HEADS + h) * (NOPE + ROPE);
#pragma unroll
    for (int c = 0; c < 4; ++c)
#pragma unroll
      for (int v = 0; v < 8; ++v) {
        const float2 f = *(const float2*)(qrow + c*32 + a_koff(v, half));
        qn[c].u32[v] = pack2bf(f.x, f.y);
      }
#pragma unroll
    for (int c = 0; c < 2; ++c)
#pragma unroll
      for (int v = 0; v < 8; ++v) {
        const float2 f = *(const float2*)(qrow + NOPE + c*32 + a_koff(v, half));
        qr[c].u32[v] = pack2bf(f.x, f.y);
      }
  }

  const v8f vz = {0.f,0.f,0.f,0.f,0.f,0.f,0.f,0.f};
  v8f oacc[8];
#pragma unroll
  for (int i = 0; i < 8; ++i) oacc[i] = vz;
  float mrow[8], lrow[8];
#pragma unroll
  for (int v = 0; v < 8; ++v) { mrow[v] = NEG_BIG; lrow[v] = 0.f; }

  const int qmax = qb + 15;
  v_prefetch(0, 0);                       // prologue: V tile for j0 = 0

  // j0 is a multiple of 32 and <= qmax <= 1023, so j0 <= 992: every key index
  // j0..j0+31 and jb+l16 (jb<=qmax) is < SEQ -- all loads in-bounds by design.
  for (int j0 = 0; j0 <= qmax; j0 += 32) {
    const int buf = (j0 >> 5) & 1;
    // Warm the cache for next tile's K rows while this tile computes.
    if (j0 + 32 <= qmax)
      __builtin_prefetch(k_ws + (size_t)(b*SEQ + j0 + 32 + lane) * (HEADS*NOPE)
                              + h*NOPE, 0, 1);

    // ---- scores for 32 keys (two 16x16 tiles) ----
    v8f sc[2];
#pragma unroll
    for (int st = 0; st < 2; ++st) {
      const int jb = j0 + st*16;
      if (jb > qmax) {
#pragma unroll
        for (int v = 0; v < 8; ++v) sc[st][v] = NEG_BIG;
        continue;
      }
      const int keyt = b * SEQ + jb + l16;
      v8f s = vz;
      const unsigned short* kcol =
          k_ws + (size_t)keyt * (HEADS*NOPE) + h*NOPE + half*16;
#pragma unroll
      for (int c = 0; c < 4; ++c) {            // q_nope . k_nope, K=128
        BFrag kb;
        kb.q4[0] = *(const uint4*)(kcol + c*32);
        kb.q4[1] = *(const uint4*)(kcol + c*32 + 8);
        s = __builtin_amdgcn_wmma_f32_16x16x32_bf16(
            false, qn[c].v, false, kb.v, (short)0, s, false, false);
      }
      const unsigned short* rcol = rope_bf + (size_t)keyt * ROPE + half*16;
#pragma unroll
      for (int c = 0; c < 2; ++c) {            // q_rope . rope_k, K=64
        BFrag rb;
        rb.q4[0] = *(const uint4*)(rcol + c*32);
        rb.q4[1] = *(const uint4*)(rcol + c*32 + 8);
        s = __builtin_amdgcn_wmma_f32_16x16x32_bf16(
            false, qr[c].v, false, rb.v, (short)0, s, false, false);
      }
      const int col = jb + l16;                // exp2-domain scale + causal mask
#pragma unroll
      for (int v = 0; v < 8; ++v) {
        const int row = qb + v + 8*half;
        sc[st][v] = (col > row) ? NEG_BIG : s[v] * SM_SCALE;
      }
    }

    // ---- online softmax (exp2 domain) over the 32-key tile ----
#pragma unroll
    for (int v = 0; v < 8; ++v) {
      float rm = fmaxf(sc[0][v], sc[1][v]);
#pragma unroll
      for (int msk = 1; msk < 16; msk <<= 1)
        rm = fmaxf(rm, __shfl_xor(rm, msk, 32));
      const float mn    = fmaxf(mrow[v], rm);
      const float p0    = exp2f(sc[0][v] - mn);
      const float p1    = exp2f(sc[1][v] - mn);
      const float alpha = exp2f(mrow[v] - mn);
      float rs = p0 + p1;
#pragma unroll
      for (int msk = 1; msk < 16; msk <<= 1)
        rs += __shfl_xor(rs, msk, 32);
      lrow[v] = lrow[v] * alpha + rs;
      mrow[v] = mn;
#pragma unroll
      for (int nt = 0; nt < 8; ++nt) oacc[nt][v] *= alpha;
      pw[(v + 8*half)*32 +      l16] = bf_bits(p0);   // C layout -> LDS bf16
      pw[(v + 8*half)*32 + 16 + l16] = bf_bits(p1);
    }
    asm volatile("s_wait_dscnt 0x0" ::: "memory");

    BFrag pa;                                  // P as A-matrix 16x32 (bf16)
    pa.q4[0] = *(const uint4*)(pw + l16*32 + half*16);
    pa.q4[1] = *(const uint4*)(pw + l16*32 + half*16 + 8);

    // Kick next tile's V copy, then wait for the current one (counted wait:
    // <=16 outstanding means the 16 ops of tile j0 have completed).
    if (j0 + 32 <= qmax) {
      v_prefetch(j0 + 32, buf ^ 1);
      asm volatile("s_wait_asynccnt 0x10" ::: "memory");
    } else {
      asm volatile("s_wait_asynccnt 0x0" ::: "memory");
    }

    // ---- P @ V : K = 32 keys, 8 output column tiles of 16, V from LDS ----
#pragma unroll
    for (int nt = 0; nt < 8; ++nt) {
      const unsigned short* slot = &vstage[wave][buf][(nt*32 + lane) * 16];
      BFrag vb;
      vb.q4[0] = *(const uint4*)(slot);
      vb.q4[1] = *(const uint4*)(slot + 8);
      oacc[nt] = __builtin_amdgcn_wmma_f32_16x16x32_bf16(
          false, pa.v, false, vb.v, (short)0, oacc[nt], false, false);
    }
  }

  // ---- epilogue: normalize and store f32 output [T,H,128] ----
#pragma unroll
  for (int v = 0; v < 8; ++v) {
    const float inv = 1.0f / lrow[v];
    const int t = b * SEQ + qb + v + 8*half;
    float* orow = out + ((size_t)t * HEADS + h) * VDIM;
#pragma unroll
    for (int nt = 0; nt < 8; ++nt)
      orow[nt*16 + l16] = oacc[nt][v] * inv;
  }
}

// ---------------------------------------------------------------------------
extern "C" void kernel_launch(void* const* d_in, const int* in_sizes, int n_in,
                              void* d_out, int out_size, void* d_ws, size_t ws_size,
                              hipStream_t stream) {
  const float* q_p    = (const float*)d_in[0];   // [T, H, 192]
  const float* ckv_p  = (const float*)d_in[1];   // [T, 512]
  const float* rk_p   = (const float*)d_in[2];   // [T, 64]
  const float* wup_p  = (const float*)d_in[3];   // [4096, 512]
  float* out_p        = (float*)d_out;           // [T, H, 128]

  // Workspace layout (bf16 ushorts):
  //   k_ws   [T, H*128]      : 8 Mi elems (16 MB)
  //   vT_ws  [B, H, 128, S]  : 8 Mi elems (16 MB)
  //   ckv_bf [T, 512]        : 2 Mi elems (4 MB)
  //   wup_bf [4096, 512]     : 2 Mi elems (4 MB)
  //   rope_bf[T, 64]         : 256 Ki elems (0.5 MB)
  unsigned short* k_ws    = (unsigned short*)d_ws;
  unsigned short* vT_ws   = k_ws   + (size_t)T_TOK * (HEADS*NOPE);
  unsigned short* ckv_bf  = vT_ws  + (size_t)BATCH * HEADS * VDIM * SEQ;
  unsigned short* wup_bf  = ckv_bf + (size_t)T_TOK * RANK;
  unsigned short* rope_bf = wup_bf + (size_t)(HEADS*(NOPE+VDIM)) * RANK;

  const int n4_ckv  = (T_TOK * RANK) / 4;
  const int n4_wup  = (HEADS*(NOPE+VDIM) * RANK) / 4;
  const int n4_rope = (T_TOK * ROPE) / 4;
  cvt_bf16<<<dim3((n4_ckv  + 255)/256), 256, 0, stream>>>(ckv_p, ckv_bf, n4_ckv);
  cvt_bf16<<<dim3((n4_wup  + 255)/256), 256, 0, stream>>>(wup_p, wup_bf, n4_wup);
  cvt_bf16<<<dim3((n4_rope + 255)/256), 256, 0, stream>>>(rk_p, rope_bf, n4_rope);

  dim3 g1((HEADS*(NOPE+VDIM))/64, T_TOK/64);
  mla_upproj<<<g1, 128, 0, stream>>>(ckv_bf, wup_bf, k_ws, vT_ws);

  dim3 g2(SEQ/64, HEADS, BATCH);
  mla_attn<<<g2, 128, 0, stream>>>(q_p, rope_bf, k_ws, vT_ws, out_p);
}
